// FFSpikingNet_62723702391148
// MI455X (gfx1250) — compile-verified, matrix operand order checked by reading, not run on hardware
//
#include <hip/hip_runtime.h>
#include <hip/hip_bf16.h>

typedef __attribute__((ext_vector_type(16))) _Float16 v16h;
typedef __attribute__((ext_vector_type(8)))  _Float16 v8h;
typedef __attribute__((ext_vector_type(8)))  float    v8f;
typedef __attribute__((ext_vector_type(4)))  unsigned int u32x4;
typedef __attribute__((ext_vector_type(8)))  int          i32x8;
typedef __attribute__((ext_vector_type(4)))  int          i32x4;

#define TSTEPS 16
#define MROWS  2560        // 10 classes * 256 batch
#define HDIM   2048
#define K1PAD  800         // 784 padded to multiple of 32
#define NCLS   10
#define BATCH  256

// LDS staging layout for layerT: per timestep a 32-row x 32-k f16 tile with a
// 40-half row stride (TDM pad: 16 DWORDs data + 4 DWORDs pad per row).
#define LROW   40
#define TSLAB  (32 * LROW)          // halfs per timestep slab (1280)
#define BUFH   (TSTEPS * TSLAB)     // halfs per k-slab buffer (20480 = 40KB)
#define KSTEPS (HDIM / 32)

#if defined(__has_builtin)
#if __has_builtin(__builtin_amdgcn_tensor_load_to_lds) && \
    __has_builtin(__builtin_amdgcn_s_wait_tensorcnt)
#define HAVE_TDM 1
#endif
#endif
#ifndef HAVE_TDM
#define HAVE_TDM 0
#endif

// ---------------------------------------------------------------------------
// WMMA fragment loader: 16-bit A/B fragment per ISA 7.12.2.
// lane<16 : row M=lane,    halves 0-7 = K[0..7],   halves 8-15 = K[16..23]
// lane>=16: row M=lane-16, halves 0-7 = K[8..15],  halves 8-15 = K[24..31]
// Caller passes base already offset by (row*stride + hi*8); two b128 loads
// (global_load_b128 or ds_load_b128 depending on address space inference).
// ---------------------------------------------------------------------------
__device__ __forceinline__ v16h load_frag(const _Float16* base) {
  v8h lo8 = *(const v8h*)(base);
  v8h hi8 = *(const v8h*)(base + 16);
  v16h r;
#pragma unroll
  for (int i = 0; i < 8; ++i) { r[i] = lo8[i]; r[i + 8] = hi8[i]; }
  return r;
}

#if HAVE_TDM
// ---------------------------------------------------------------------------
// One TDM descriptor stages ALL 16 timestep tiles (iterate_enable):
//   tile: 32 k-elements (row) x 32 m-rows, data_size=2B
//   row pad: 16 DWORDs data (pad_interval=3) + 4 DWORDs pad (pad_amount=3)
//   iteration: +MROWS*HDIM elements global, +TSLAB elements LDS, 16 iters
// ---------------------------------------------------------------------------
__device__ __forceinline__ void tdm_stage_A(const _Float16* S, unsigned lds_byte,
                                            size_t gelem) {
  unsigned long long ga = (unsigned long long)(const void*)(S + gelem);
  u32x4 g0;
  g0[0] = 1u;                                            // count=1, user mode
  g0[1] = lds_byte;                                      // lds_addr
  g0[2] = (unsigned)(ga & 0xFFFFFFFFull);                // global_addr lo
  g0[3] = (unsigned)((ga >> 32) & 0x1FFFFFFull) | (2u << 30);  // hi | type=2
  i32x8 g1;
  g1[0] = (1 << 16)      // data_size = 2B
        | (1 << 19)      // iterate_enable
        | (1 << 20)      // pad_enable
        | (3 << 22)      // pad_interval: 16 DWORDs
        | (3 << 25);     // pad_amount: 4 DWORDs
  g1[1] = (int)((unsigned)HDIM << 16);    // tensor_dim0[15:0]
  g1[2] = (int)((unsigned)MROWS << 16);   // tensor_dim0[31:16]=0 | tensor_dim1[15:0]
  g1[3] = (int)(32u << 16);               // tensor_dim1[31:16]=0 | tile_dim0=32
  g1[4] = 32;                             // tile_dim1=32 | tile_dim2=0
  g1[5] = HDIM;                           // tensor_dim0_stride[31:0]
  g1[6] = 0;                              // stride hi | dim1_stride lo
  g1[7] = 0;
  i32x4 g2;
  g2[0] = 0;                              // tensor_dim2 (unused)
  g2[1] = TSLAB;                          // lds_addr_increment per iteration
  g2[2] = (int)(MROWS * HDIM);            // global_addr_increment per iteration
  g2[3] = (15 << 16);                     // iterate_count = 15 -> 16 iterations
  i32x4 g3 = {0, 0, 0, 0};
  i32x8 g4 = {0, 0, 0, 0, 0, 0, 0, 0};    // extra group (6-arg toolchain)
  __builtin_amdgcn_tensor_load_to_lds(g0, g1, g2, g3, g4, 0);
}
#endif

// ---------------------------------------------------------------------------
// Build label-superimposed input x_all [2560][800] in f16 (zero-padded K).
// ---------------------------------------------------------------------------
__global__ void prep_xall_kernel(const float* __restrict__ x,
                                 _Float16* __restrict__ xall) {
  const int row = blockIdx.x;          // 0..2559
  const int c   = row / BATCH;
  const int b   = row % BATCH;
  const int tid = threadIdx.x;
  __shared__ float smax[256];
  float m = 0.0f;                       // pixels are in [0,1]
  for (int k = tid; k < 784; k += 256) {
    float v = (k < NCLS) ? 0.0f : x[b * 784 + k];
    m = fmaxf(m, v);
  }
  smax[tid] = m;
  __syncthreads();
  for (int s = 128; s > 0; s >>= 1) {
    if (tid < s) smax[tid] = fmaxf(smax[tid], smax[tid + s]);
    __syncthreads();
  }
  const float vmax = smax[0] + 1e-6f;
  for (int k = tid; k < K1PAD; k += 256) {
    float v;
    if (k >= 784)      v = 0.0f;
    else if (k == c)   v = vmax;
    else if (k < NCLS) v = 0.0f;
    else               v = x[b * 784 + k];
    xall[(size_t)row * K1PAD + k] = (_Float16)v;
  }
}

__global__ void conv_w1_kernel(const float* __restrict__ w,
                               _Float16* __restrict__ o) {
  int idx = blockIdx.x * 256 + threadIdx.x;   // 2048*800 exact
  int h = idx / K1PAD, k = idx % K1PAD;
  o[idx] = (_Float16)((k < 784) ? w[h * 784 + k] : 0.0f);
}

__global__ void conv_w_kernel(const float* __restrict__ w,
                              _Float16* __restrict__ o, int n) {
  int idx = blockIdx.x * 256 + threadIdx.x;
  if (idx < n) o[idx] = (_Float16)w[idx];
}

__global__ void zero_f32_kernel(float* __restrict__ p, int n) {
  int idx = blockIdx.x * 256 + threadIdx.x;
  if (idx < n) p[idx] = 0.0f;
}

// ---------------------------------------------------------------------------
// Layer 1: static input over time -> single GEMM (K=800) + 16-step LIF on a
// constant current, all in registers. One wave = one 16x16 tile.
// ---------------------------------------------------------------------------
__global__ void layer1_kernel(const _Float16* __restrict__ A,
                              const _Float16* __restrict__ W,
                              const float* __restrict__ bias,
                              _Float16* __restrict__ spk,
                              float* __restrict__ gsum) {
  const int tid  = threadIdx.x;
  const int wave = tid >> 5;
  const int lane = tid & 31;
  const int lo   = lane & 15;
  const int hi   = lane >> 4;
  const int m0   = blockIdx.x * 32 + (wave >> 2) * 16;
  const int h0   = blockIdx.y * 64 + (wave & 3) * 16;

  const _Float16* arow = A + (size_t)(m0 + lo) * K1PAD + hi * 8;
  const _Float16* brow = W + (size_t)(h0 + lo) * K1PAD + hi * 8;

  v8f acc = {};
  for (int k = 0; k < K1PAD; k += 32) {
    v16h a = load_frag(arow + k);
    v16h b = load_frag(brow + k);
    acc = __builtin_amdgcn_wmma_f32_16x16x32_f16(
        false, a, false, b, (short)0, acc, false, false);
  }

  const float bh = bias[h0 + lo];
  const int   n  = h0 + lo;
  float cur[8], v[8], cnt[8];
#pragma unroll
  for (int r = 0; r < 8; ++r) { cur[r] = acc[r] + bh; v[r] = 0.f; cnt[r] = 0.f; }

  const size_t tstride = (size_t)MROWS * HDIM;
#pragma unroll
  for (int t = 0; t < TSTEPS; ++t) {
#pragma unroll
    for (int r = 0; r < 8; ++r) {
      float vv = (v[r] + cur[r]) * 0.5f;          // v += (x - v)/TAU, TAU=2
      float s  = (vv >= 1.0f) ? 1.0f : 0.0f;      // spike(v - V_TH)
      cnt[r] += s;
      v[r]   = vv * (1.0f - s);                   // hard reset to 0
      spk[(size_t)t * tstride + (size_t)(m0 + hi * 8 + r) * HDIM + n] =
          (_Float16)s;
    }
  }
#pragma unroll
  for (int r = 0; r < 8; ++r)
    atomicAdd(&gsum[m0 + hi * 8 + r], cnt[r] * cnt[r] * (1.0f / (float)HDIM));
}

// ---------------------------------------------------------------------------
// Layers 2/3: 16 timestep GEMMs fused. TDM double-buffer stages the 16
// timestep A tiles (32 rows x 32 k) into LDS once per block per k-step;
// all 8 waves read A fragments with ds_load_b128. B fragments stay global
// (L0-resident). 16 f32 accumulators per wave, then in-register LIF.
// ---------------------------------------------------------------------------
extern __shared__ _Float16 smem[];

__global__ void layerT_kernel(const _Float16* __restrict__ S,
                              const _Float16* __restrict__ W,
                              const float* __restrict__ bias,
                              _Float16* __restrict__ spk_out,
                              float* __restrict__ gsum,
                              int write_spk) {
  const int tid  = threadIdx.x;
  const int wave = tid >> 5;
  const int lane = tid & 31;
  const int lo   = lane & 15;
  const int hi   = lane >> 4;
  const int m0b  = blockIdx.x * 32;                 // block's 32-row m tile
  const int wm   = wave >> 2;                       // 0..1
  const int m0   = m0b + wm * 16;
  const int h0   = blockIdx.y * 64 + (wave & 3) * 16;

  const size_t    tstride = (size_t)MROWS * HDIM;
  const _Float16* brow = W + (size_t)(h0 + lo) * HDIM + hi * 8;

  v8f acc[TSTEPS];
#pragma unroll
  for (int t = 0; t < TSTEPS; ++t) acc[t] = (v8f){};

#if HAVE_TDM
  const unsigned lds_base = (unsigned)(size_t)(void*)smem;
  if (wave == 0) tdm_stage_A(S, lds_base, (size_t)m0b * HDIM);
#endif

  for (int ks = 0; ks < KSTEPS; ++ks) {
    const int k = ks * 32;
    _Float16* buf = smem + (size_t)(ks & 1) * BUFH;

#if HAVE_TDM
    if (wave == 0) {
      if (ks + 1 < KSTEPS) {
        tdm_stage_A(S, lds_base + (unsigned)(((ks + 1) & 1) * BUFH * 2),
                    (size_t)m0b * HDIM + (size_t)(k + 32));
        __builtin_amdgcn_s_wait_tensorcnt(1);   // oldest (current slab) done
      } else {
        __builtin_amdgcn_s_wait_tensorcnt(0);
      }
    }
    __syncthreads();
#else
    // synchronous staging fallback: 2048 16B units, 8 per thread
    for (int u = tid; u < TSTEPS * 32 * 4; u += 256) {
      const int t = u >> 7, rem = u & 127, row = rem >> 2, seg = rem & 3;
      *(v8h*)&buf[t * TSLAB + row * LROW + seg * 8] =
          *(const v8h*)(S + (size_t)t * tstride +
                        (size_t)(m0b + row) * HDIM + k + seg * 8);
    }
    __syncthreads();
#endif

    v16h b = load_frag(brow + k);
    const _Float16* abase = buf + (wm * 16 + lo) * LROW + hi * 8;
#pragma unroll
    for (int t = 0; t < TSTEPS; ++t) {
      v16h a = load_frag(abase + t * TSLAB);
      acc[t] = __builtin_amdgcn_wmma_f32_16x16x32_f16(
          false, a, false, b, (short)0, acc[t], false, false);
    }
    __syncthreads();   // all reads of buf done before it is restaged
  }

  const float bh = bias[h0 + lo];
  const int   n  = h0 + lo;
  float v[8], cnt[8];
#pragma unroll
  for (int r = 0; r < 8; ++r) { v[r] = 0.f; cnt[r] = 0.f; }

#pragma unroll
  for (int t = 0; t < TSTEPS; ++t) {
#pragma unroll
    for (int r = 0; r < 8; ++r) {
      float c  = acc[t][r] + bh;
      float vv = (v[r] + c) * 0.5f;
      float s  = (vv >= 1.0f) ? 1.0f : 0.0f;
      cnt[r] += s;
      v[r]   = vv * (1.0f - s);
      if (write_spk)
        spk_out[(size_t)t * tstride + (size_t)(m0 + hi * 8 + r) * HDIM + n] =
            (_Float16)s;
    }
  }
#pragma unroll
  for (int r = 0; r < 8; ++r)
    atomicAdd(&gsum[m0 + hi * 8 + r], cnt[r] * cnt[r] * (1.0f / (float)HDIM));
}

// gsum[c*256+b] -> gpc[b][c] (2560 floats) then pred[b] (256 floats)
__global__ void finalize_kernel(const float* __restrict__ gsum,
                                float* __restrict__ out) {
  const int b = threadIdx.x;   // 256 threads
  float best = -1e30f;
  int   bc   = 0;
#pragma unroll
  for (int c = 0; c < NCLS; ++c) {
    float g = gsum[c * BATCH + b];
    out[b * NCLS + c] = g;
    if (g > best) { best = g; bc = c; }   // strict > keeps first max (argmax)
  }
  out[NCLS * BATCH + b] = (float)bc;
}

// ---------------------------------------------------------------------------
extern "C" void kernel_launch(void* const* d_in, const int* in_sizes, int n_in,
                              void* d_out, int out_size, void* d_ws,
                              size_t ws_size, hipStream_t stream) {
  (void)in_sizes; (void)n_in; (void)out_size; (void)ws_size;
  const float* x  = (const float*)d_in[0];
  const float* W1 = (const float*)d_in[1];
  const float* b1 = (const float*)d_in[2];
  const float* W2 = (const float*)d_in[3];
  const float* b2 = (const float*)d_in[4];
  const float* W3 = (const float*)d_in[5];
  const float* b3 = (const float*)d_in[6];

  char* ws = (char*)d_ws;
  size_t off = 0;
  _Float16* xall = (_Float16*)(ws + off); off += (size_t)MROWS * K1PAD * 2;   // 4.10 MB
  _Float16* w1f  = (_Float16*)(ws + off); off += (size_t)HDIM * K1PAD * 2;    // 3.28 MB
  _Float16* w2f  = (_Float16*)(ws + off); off += (size_t)HDIM * HDIM * 2;     // 8.39 MB
  _Float16* w3f  = (_Float16*)(ws + off); off += (size_t)HDIM * HDIM * 2;     // 8.39 MB
  float*    gsum = (float*)(ws + off);    off += (size_t)MROWS * 4;           // 10 KB
  off = (off + 255) & ~(size_t)255;
  _Float16* spkA = (_Float16*)(ws + off); off += (size_t)TSTEPS * MROWS * HDIM * 2; // 168 MB
  _Float16* spkB = (_Float16*)(ws + off); off += (size_t)TSTEPS * MROWS * HDIM * 2; // 168 MB

  // prep
  prep_xall_kernel<<<MROWS, 256, 0, stream>>>(x, xall);
  conv_w1_kernel<<<(HDIM * K1PAD) / 256, 256, 0, stream>>>(W1, w1f);
  conv_w_kernel<<<(HDIM * HDIM) / 256, 256, 0, stream>>>(W2, w2f, HDIM * HDIM);
  conv_w_kernel<<<(HDIM * HDIM) / 256, 256, 0, stream>>>(W3, w3f, HDIM * HDIM);
  zero_f32_kernel<<<(MROWS + 255) / 256, 256, 0, stream>>>(gsum, MROWS);

  // layers
  dim3 grid(MROWS / 32, HDIM / 64);         // 80 x 32 blocks, 8 waves each
  const size_t lds_bytes = 2u * BUFH * 2u;  // 80 KB double buffer
  layer1_kernel<<<grid, 256, 0, stream>>>(xall, w1f, b1, spkA, gsum);
  layerT_kernel<<<grid, 256, lds_bytes, stream>>>(spkA, w2f, b2, spkB, gsum, 1);
  layerT_kernel<<<grid, 256, lds_bytes, stream>>>(spkB, w3f, b3, spkA, gsum, 0);

  // outputs: gpc [256,10] then pred [256] (as float)
  finalize_kernel<<<1, 256, 0, stream>>>(gsum, (float*)d_out);
}
